// Attention_4501125726440
// MI455X (gfx1250) — matrix operand order repaired and checked by reading, not compiled, not run on hardware
//
#include <hip/hip_runtime.h>
#include <hip/hip_bf16.h>

// ---------------------------------------------------------------------------
// MI455X / gfx1250 attention layer: one-shot f32->bf16 convert, then
// QKV(+RMSNorm+RoPE) -> flash attn w/ sink -> output projection.
// GEMMs: 64x64 wave tiles, v_wmma_f32_16x16x32_bf16 (wave32),
// 16 b128 loads : 16 WMMAs per k-step (1:1), epilogues in C-layout registers.
// ---------------------------------------------------------------------------

#define S_LEN 2048
#define DMODEL 1024
#define NHEAD 16
#define HDIM 64
#define BSZ 2
#define MTOT (BSZ * S_LEN)          // 4096 rows
#define LN10000_OVER_32 0.28782313662425576f   // ln(10000)/32

typedef __attribute__((ext_vector_type(16))) __bf16 v16bf;
typedef __attribute__((ext_vector_type(8)))  float  v8f;

union ABu {
  v16bf v;
  unsigned short u[16];
  uint4 q[2];
};

__device__ __forceinline__ v8f zero8() {
  v8f z = {0.f, 0.f, 0.f, 0.f, 0.f, 0.f, 0.f, 0.f};
  return z;
}

__device__ __forceinline__ unsigned short bfbits(float f) {
  union { float f; unsigned int u; } a; a.f = f;
  unsigned int u = a.u + 0x7FFFu + ((a.u >> 16) & 1u);   // RNE
  return (unsigned short)(u >> 16);
}

__device__ __forceinline__ v8f wmma_bf16(const ABu& a, const ABu& b, v8f c) {
  return __builtin_amdgcn_wmma_f32_16x16x32_bf16(
      /*neg_a=*/false, a.v, /*neg_b=*/false, b.v,
      /*c_mod=*/(short)0, c, /*reuse_a=*/false, /*reuse_b=*/false);
}

// ---- A operand: 16xK row-major tile. Lane holds row M=lane%16.
//      K pattern per lane: {kb..kb+7, kb+16..kb+23}, kb = 8*(lane>>4).
__device__ __forceinline__ void loadA_bf(ABu& t, const unsigned short* rp, int kb) {
  t.q[0] = *reinterpret_cast<const uint4*>(rp + kb);
  t.q[1] = *reinterpret_cast<const uint4*>(rp + kb + 16);
}

// ---- B operand: source is row-major NxK tile (= B^T). Lane holds col N=lane%16.
//      K pattern per lane: {ko/2..ko/2+7, ko/2+16..ko/2+23}, ko = 16*(lane>>4).
__device__ __forceinline__ void loadB_bf(ABu& t, const unsigned short* rp, int ko) {
  t.q[0] = *reinterpret_cast<const uint4*>(rp + (ko >> 1));
  t.q[1] = *reinterpret_cast<const uint4*>(rp + (ko >> 1) + 16);
}

__device__ __forceinline__ float redmax16(float v) {
  v = fmaxf(v, __shfl_xor(v, 1, 16));
  v = fmaxf(v, __shfl_xor(v, 2, 16));
  v = fmaxf(v, __shfl_xor(v, 4, 16));
  v = fmaxf(v, __shfl_xor(v, 8, 16));
  return v;
}
__device__ __forceinline__ float redsum16(float v) {
  v += __shfl_xor(v, 1, 16);
  v += __shfl_xor(v, 2, 16);
  v += __shfl_xor(v, 4, 16);
  v += __shfl_xor(v, 8, 16);
  return v;
}

// ---- 64x64x32 GEMM k-step: 8 A-loads + 8 B-loads feed 16 WMMAs (1:1) ----
__device__ __forceinline__ void gemm_load64(
    ABu a[4], ABu bt[4], const unsigned short* abase,
    const unsigned short* brow0, int kk, int kbA, int koB) {
#pragma unroll
  for (int g = 0; g < 4; g++)
    loadA_bf(a[g], abase + (size_t)(g * 16) * DMODEL + kk, kbA);
#pragma unroll
  for (int n = 0; n < 4; n++)
    loadB_bf(bt[n], brow0 + (size_t)(n * 16) * DMODEL + kk, koB);
}
__device__ __forceinline__ void gemm_compute64(v8f c[4][4], ABu a[4], ABu bt[4]) {
#pragma unroll
  for (int n = 0; n < 4; n++)
#pragma unroll
    for (int g = 0; g < 4; g++)
      c[g][n] = wmma_bf16(a[g], bt[n], c[g][n]);
}

// ===========================================================================
// Stage 0: bulk f32 -> bf16 conversion (memory bound, ~2us at 23.3 TB/s).
// ===========================================================================
__global__ __launch_bounds__(256) void cvt_kernel(
    const float* __restrict__ src, unsigned short* __restrict__ dst) {
  const size_t i = ((size_t)blockIdx.x * 256 + threadIdx.x) * 8;
  float4 a = *reinterpret_cast<const float4*>(src + i);
  float4 b = *reinterpret_cast<const float4*>(src + i + 4);
  union { unsigned short u[8]; uint4 q; } o;
  o.u[0] = bfbits(a.x); o.u[1] = bfbits(a.y); o.u[2] = bfbits(a.z); o.u[3] = bfbits(a.w);
  o.u[4] = bfbits(b.x); o.u[5] = bfbits(b.y); o.u[6] = bfbits(b.z); o.u[7] = bfbits(b.w);
  *reinterpret_cast<uint4*>(dst + i) = o.q;
}

// ===========================================================================
// Stage 1: fused QKV projection + per-head RMSNorm + RoPE.
// blockIdx.z = mode: 0=Q, 1=K, 2=V.  Per wave: 64 rows x 1 head (64 cols).
// Q,K -> bf16 [B,H,S,HD];  V -> bf16 transposed [B,H,HD,S].
// ===========================================================================
__global__ __launch_bounds__(256) void qkv_kernel(
    const unsigned short* __restrict__ xbf,
    const unsigned short* __restrict__ wqbf,
    const unsigned short* __restrict__ wkbf,
    const unsigned short* __restrict__ wvbf,
    const float* __restrict__ qnw, const float* __restrict__ knw,
    unsigned short* __restrict__ Qbf, unsigned short* __restrict__ Kbf,
    unsigned short* __restrict__ Vt) {
  const int lane   = threadIdx.x & 31;
  const int wave   = threadIdx.x >> 5;
  const int lanelo = lane & 15;
  const int half   = lane >> 4;
  const int kbA    = half * 8;
  const int koB    = half * 16;

  const int tile = blockIdx.x * 8 + wave;   // 0..1023
  const int mt   = tile >> 4;               // 0..63 (64-row tiles)
  const int nt   = tile & 15;               // head index
  const int mode = blockIdx.z;

  const unsigned short* W = (mode == 0) ? wqbf : ((mode == 1) ? wkbf : wvbf);

  v8f c[4][4];
#pragma unroll
  for (int g = 0; g < 4; g++)
#pragma unroll
    for (int n = 0; n < 4; n++) c[g][n] = zero8();

  const unsigned short* abase = xbf + (size_t)(mt * 64 + lanelo) * DMODEL;
  const unsigned short* brow0 = W + (size_t)(nt * 64 + lanelo) * DMODEL;

  for (int kk = 0; kk < DMODEL; kk += 32) {
    ABu a[4], bt[4];
    gemm_load64(a, bt, abase, brow0, kk, kbA, koB);
    gemm_compute64(c, a, bt);
  }

  // ---- epilogue in accumulator layout: row = r + 8*half, col = n*16+lanelo
  float invf[2];
  invf[0] = __expf(-(float)lanelo * LN10000_OVER_32);          // d%32 = lanelo
  invf[1] = __expf(-(float)(16 + lanelo) * LN10000_OVER_32);   // d%32 = 16+lanelo
#pragma unroll
  for (int g = 0; g < 4; g++) {
    if (mode < 2) {
      const float* nw = (mode == 0) ? qnw : knw;
      float yn[4][8];
#pragma unroll
      for (int r = 0; r < 8; r++) {
        float ss = 0.f;
#pragma unroll
        for (int n = 0; n < 4; n++) { float v = c[g][n][r]; ss += v * v; }
        ss = redsum16(ss);
        float rs = rsqrtf(ss * (1.0f / 64.0f) + 1e-6f);
#pragma unroll
        for (int n = 0; n < 4; n++)
          yn[n][r] = c[g][n][r] * rs * nw[n * 16 + lanelo];
      }
      // RoPE: rot(d) = -y[d+32] (d<32), +y[d-32] (d>=32); partner is subtile +-2.
      // Only two unique angles per lane per row (d%32 = lanelo or 16+lanelo).
#pragma unroll
      for (int r = 0; r < 8; r++) {
        const int row  = mt * 64 + g * 16 + r + half * 8;
        const int spos = row & (S_LEN - 1);
        float sn01[2], cs01[2];
        __sincosf((float)spos * invf[0], &sn01[0], &cs01[0]);
        __sincosf((float)spos * invf[1], &sn01[1], &cs01[1]);
#pragma unroll
        for (int n = 0; n < 4; n++) {
          const float cs = cs01[n & 1];
          const float sn = sn01[n & 1];
          const float rot = (n < 2) ? -yn[n + 2][r] : yn[n - 2][r];
          c[g][n][r] = yn[n][r] * cs + rot * sn;
        }
      }
    }
#pragma unroll
    for (int r = 0; r < 8; r++) {
      const int row  = mt * 64 + g * 16 + r + half * 8;
      const int bb   = row >> 11;
      const int spos = row & (S_LEN - 1);
      const int bh   = bb * NHEAD + nt;
#pragma unroll
      for (int n = 0; n < 4; n++) {
        const int d = n * 16 + lanelo;
        const unsigned short v = bfbits(c[g][n][r]);
        if (mode == 0)      Qbf[((size_t)bh * S_LEN + spos) * HDIM + d] = v;
        else if (mode == 1) Kbf[((size_t)bh * S_LEN + spos) * HDIM + d] = v;
        else                Vt[((size_t)bh * HDIM + d) * S_LEN + spos]  = v;
      }
    }
  }
}

// ===========================================================================
// Stage 2: causal flash attention with per-head sink, online softmax.
// One wave per (b, h, 16-query tile); key tiles of 32 (one WMMA K).
// K and V operands are loaded together so V fetch overlaps the score WMMAs
// and softmax VALU; P transposed C->A layout through per-wave LDS.
// ===========================================================================
__global__ __launch_bounds__(128) void attn_kernel(
    const unsigned short* __restrict__ Qbf,
    const unsigned short* __restrict__ Kbf,
    const unsigned short* __restrict__ Vt,
    const float* __restrict__ sink,
    unsigned short* __restrict__ Obf) {
  __shared__ __align__(16) unsigned short Pst[4][16 * 32];

  const int lane   = threadIdx.x & 31;
  const int wave   = threadIdx.x >> 5;
  const int lanelo = lane & 15;
  const int half   = lane >> 4;
  const int kbA    = half * 8;
  const int koB    = half * 16;

  const int id = blockIdx.x * 4 + wave;     // 0..4095
  const int qt = id & 127;
  const int h  = (id >> 7) & (NHEAD - 1);
  const int b  = id >> 11;
  const int bh = b * NHEAD + h;

  const unsigned short* Qp = Qbf + (size_t)bh * S_LEN * HDIM;
  const unsigned short* Kp = Kbf + (size_t)bh * S_LEN * HDIM;
  const unsigned short* Vp = Vt  + (size_t)bh * HDIM * S_LEN;
  unsigned short* Pw = Pst[wave];

  // Q tile (A operand), two K=32 chunks of HD=64
  ABu qa0, qa1;
  {
    const unsigned short* qrow = Qp + (size_t)(qt * 16 + lanelo) * HDIM;
    loadA_bf(qa0, qrow, kbA);
    loadA_bf(qa1, qrow + 32, kbA);
  }

  float mrow[8], lrow[8];
  const float slog = sink[h];
#pragma unroll
  for (int r = 0; r < 8; r++) { mrow[r] = slog; lrow[r] = 1.0f; }
  v8f o[4];
#pragma unroll
  for (int n = 0; n < 4; n++) o[n] = zero8();

  const int nkt = (qt * 16 + 15) / 32 + 1;
  const int q0  = qt * 16 + half * 8;

  for (int kt = 0; kt < nkt; kt++) {
    // ---- issue ALL loads for this tile first: K (4) then V (4).
    ABu kb0, kb1, kb2, kb3, vb[4];
    {
      const unsigned short* kr = Kp + (size_t)(kt * 32 + lanelo) * HDIM;
      loadB_bf(kb0, kr, koB);
      loadB_bf(kb1, kr + 32, koB);
      const unsigned short* kr2 = Kp + (size_t)(kt * 32 + 16 + lanelo) * HDIM;
      loadB_bf(kb2, kr2, koB);
      loadB_bf(kb3, kr2 + 32, koB);
#pragma unroll
      for (int n = 0; n < 4; n++)
        loadB_bf(vb[n], Vp + (size_t)(n * 16 + lanelo) * S_LEN + kt * 32, koB);
    }

    // ---- scores: 16q x 32k = two N-subtiles (V loads still in flight)
    v8f s0 = zero8(), s1 = zero8();
    s0 = wmma_bf16(qa0, kb0, s0);
    s0 = wmma_bf16(qa1, kb1, s0);
    s1 = wmma_bf16(qa0, kb2, s1);
    s1 = wmma_bf16(qa1, kb3, s1);

    const int kg0 = kt * 32 + lanelo;
    const int kg1 = kg0 + 16;
#pragma unroll
    for (int r = 0; r < 8; r++) {
      const int qg = q0 + r;
      float a = s0[r] * 0.125f;               // HD^-0.5
      float bb = s1[r] * 0.125f;
      if (kg0 > qg) a  = -1e30f;              // causal mask
      if (kg1 > qg) bb = -1e30f;
      float mx   = redmax16(fmaxf(a, bb));
      float mnew = fmaxf(mrow[r], mx);
      float alpha = __expf(mrow[r] - mnew);
      float p0 = __expf(a - mnew);
      float p1 = __expf(bb - mnew);
      lrow[r] = lrow[r] * alpha + redsum16(p0 + p1);
      mrow[r] = mnew;
#pragma unroll
      for (int n = 0; n < 4; n++) o[n][r] *= alpha;
      // stage P (C-layout) into LDS row-major 16x32
      Pw[(r + half * 8) * 32 + lanelo]      = bfbits(p0);
      Pw[(r + half * 8) * 32 + 16 + lanelo] = bfbits(p1);
    }
    // cross-lane LDS visibility within the wave (LDS is in-order per wave)
    asm volatile("s_wait_dscnt 0" ::: "memory");

    // reload P in A-operand layout
    ABu pa;
    {
      const unsigned short* prow = Pw + lanelo * 32;
      pa.q[0] = *reinterpret_cast<const uint4*>(prow + kbA);
      pa.q[1] = *reinterpret_cast<const uint4*>(prow + kbA + 16);
    }
    // out += P(16x32) x V(32xHD)
#pragma unroll
    for (int n = 0; n < 4; n++) o[n] = wmma_bf16(pa, vb[n], o[n]);
  }

  // ---- normalize and store to O [B,S, H*64] (bf16)
#pragma unroll
  for (int r = 0; r < 8; r++) {
    const float inv = 1.0f / lrow[r];
    const int row = qt * 16 + r + half * 8;
#pragma unroll
    for (int n = 0; n < 4; n++) {
      Obf[((size_t)b * S_LEN + row) * DMODEL + h * 64 + n * 16 + lanelo] =
          bfbits(o[n][r] * inv);
    }
  }
}

// ===========================================================================
// Stage 3: output projection  out = O @ wo^T  (M=4096, N=1024, K=1024), f32 out
// ===========================================================================
__global__ __launch_bounds__(256) void proj_kernel(
    const unsigned short* __restrict__ Obf,
    const unsigned short* __restrict__ wobf,
    float* __restrict__ out) {
  const int lane   = threadIdx.x & 31;
  const int wave   = threadIdx.x >> 5;
  const int lanelo = lane & 15;
  const int half   = lane >> 4;
  const int kbA    = half * 8;
  const int koB    = half * 16;

  const int tile = blockIdx.x * 8 + wave;   // 0..1023
  const int mt   = tile >> 4;               // 0..63
  const int nt   = tile & 15;

  v8f c[4][4];
#pragma unroll
  for (int g = 0; g < 4; g++)
#pragma unroll
    for (int n = 0; n < 4; n++) c[g][n] = zero8();

  const unsigned short* abase = Obf + (size_t)(mt * 64 + lanelo) * DMODEL;
  const unsigned short* brow0 = wobf + (size_t)(nt * 64 + lanelo) * DMODEL;

  for (int kk = 0; kk < DMODEL; kk += 32) {
    ABu a[4], bt[4];
    gemm_load64(a, bt, abase, brow0, kk, kbA, koB);
    gemm_compute64(c, a, bt);
  }

#pragma unroll
  for (int g = 0; g < 4; g++)
#pragma unroll
    for (int r = 0; r < 8; r++) {
      const int row = mt * 64 + g * 16 + r + half * 8;
#pragma unroll
      for (int n = 0; n < 4; n++)
        out[(size_t)row * DMODEL + nt * 64 + n * 16 + lanelo] = c[g][n][r];
    }
}

// ===========================================================================
extern "C" void kernel_launch(void* const* d_in, const int* in_sizes, int n_in,
                              void* d_out, int out_size, void* d_ws, size_t ws_size,
                              hipStream_t stream) {
  (void)in_sizes; (void)n_in; (void)out_size; (void)ws_size;
  const float* x    = (const float*)d_in[0];
  const float* wq   = (const float*)d_in[1];
  const float* wk   = (const float*)d_in[2];
  const float* wv   = (const float*)d_in[3];
  const float* wo   = (const float*)d_in[4];
  const float* qnw  = (const float*)d_in[5];
  const float* knw  = (const float*)d_in[6];
  const float* sink = (const float*)d_in[7];
  float* out = (float*)d_out;

  // workspace layout (bf16 = unsigned short), needs ~48 MB
  const size_t N  = (size_t)BSZ * NHEAD * S_LEN * HDIM;   // 4,194,304
  const size_t NW = (size_t)DMODEL * DMODEL;              // 1,048,576
  unsigned short* Qbf  = (unsigned short*)d_ws;
  unsigned short* Kbf  = Qbf + N;
  unsigned short* Vt   = Kbf + N;
  unsigned short* Obf  = Vt + N;
  unsigned short* xbf  = Obf + N;          // 4,194,304 (= MTOT*DMODEL)
  unsigned short* wqbf = xbf + N;
  unsigned short* wkbf = wqbf + NW;
  unsigned short* wvbf = wkbf + NW;
  unsigned short* wobf = wvbf + NW;

  // Stage 0: bulk conversions (grid = elems / (256*8))
  cvt_kernel<<<2048, 256, 0, stream>>>(x, xbf);
  cvt_kernel<<<512, 256, 0, stream>>>(wq, wqbf);
  cvt_kernel<<<512, 256, 0, stream>>>(wk, wkbf);
  cvt_kernel<<<512, 256, 0, stream>>>(wv, wvbf);
  cvt_kernel<<<512, 256, 0, stream>>>(wo, wobf);

  qkv_kernel<<<dim3(128, 1, 3), 256, 0, stream>>>(xbf, wqbf, wkbf, wvbf,
                                                  qnw, knw, Qbf, Kbf, Vt);
  attn_kernel<<<1024, 128, 0, stream>>>(Qbf, Kbf, Vt, sink, Obf);
  proj_kernel<<<128, 256, 0, stream>>>(Obf, wobf, out);
}
